// LatentSDE_9964324127213
// MI455X (gfx1250) — compile-verified, matrix operand order checked by reading, not run on hardware
//
#include <hip/hip_runtime.h>
#include <cstdint>
#include <cstddef>

// ---------------------------------------------------------------------------
// Types for CDNA5 WMMA (wave32, 16x16x32 bf16 -> f32)
// ---------------------------------------------------------------------------
typedef __bf16 bf16_t;
typedef __attribute__((ext_vector_type(16))) __bf16 v16bf;
typedef __attribute__((ext_vector_type(8)))  __bf16 v8bf;
typedef __attribute__((ext_vector_type(8)))  float  v8f;

#define T_      50
#define B_      2048
#define D_      128
#define L_      16
#define C_      64
#define H_      128
#define A_      8
#define SKIP_   5
#define NSEG_   10
#define NSTEPS_ 8
#define DT_     0.25f
#define SQRTDT_ 0.5f
#define LOG2PI_ 1.8378770664093453f

#define NWAVES  8   // 256 threads / wave32

__device__ __forceinline__ bf16_t f2bf(float f) {
  union { float f; unsigned u; } a; a.f = f;
  unsigned r = (a.u + 0x7FFFu + ((a.u >> 16) & 1u)) >> 16;
  union { unsigned short s; bf16_t b; } c; c.s = (unsigned short)r;
  return c.b;
}
__device__ __forceinline__ float sp(float x)   { return x > 20.f ? x : log1pf(expf(x)); }
__device__ __forceinline__ float sigm(float x) { return 1.f / (1.f + expf(-x)); }

// ---------------------------------------------------------------------------
// Async global->LDS staging (CDNA5 ASYNCcnt path)
// ---------------------------------------------------------------------------
__device__ __forceinline__ void async_ld16(unsigned lds_off, const void* gaddr) {
  asm volatile("global_load_async_to_lds_b128 %0, %1, off"
               :: "v"(lds_off), "v"(gaddr) : "memory");
}
__device__ __forceinline__ void wait_async0() {
  asm volatile("s_wait_asynccnt 0" ::: "memory");
}

// ---------------------------------------------------------------------------
// WMMA GEMM over LDS A (bf16) x fragment-packed global B (bf16) -> LDS out.
// A layout per 16-bit A-matrix spec: lanes 0-15 rows 0-15 hold K 0-7,16-23;
// lanes 16-31 hold K 8-15,24-31. B tiles are pre-packed so each lane does one
// contiguous 32-byte load. ACT: 0=none 1=softplus 2=tanh. OBF: bf16 output.
// ---------------------------------------------------------------------------
template<int MT, int NT, int KT, int ACT, int OBF>
__device__ __forceinline__ void gemm_lds(const bf16_t* __restrict__ Amat, int sa,
                                         const bf16_t* __restrict__ Bp,
                                         const float*  __restrict__ bias,
                                         void* __restrict__ outp, int so,
                                         int wid, int lane)
{
  for (int t = wid; t < MT * NT; t += NWAVES) {
    int mt = t % MT, nt = t / MT;
    v8f acc = {0.f,0.f,0.f,0.f,0.f,0.f,0.f,0.f};
    const bf16_t* arow = Amat + (size_t)mt * 16 * sa + (lane & 15) * sa + ((lane & 16) ? 8 : 0);
    const bf16_t* bt   = Bp + (size_t)nt * KT * 512 + lane * 16;
#pragma unroll
    for (int kt = 0; kt < KT; ++kt) {
      union { v16bf v; v8bf h[2]; } ua;
      ua.h[0] = *(const v8bf*)(arow + kt * 32);
      ua.h[1] = *(const v8bf*)(arow + kt * 32 + 16);
      v16bf bv = *(const v16bf*)(bt + kt * 512);
      acc = __builtin_amdgcn_wmma_f32_16x16x32_bf16(false, ua.v, false, bv,
                                                    (short)0, acc, false, false);
    }
    int col  = nt * 16 + (lane & 15);
    int row0 = mt * 16 + ((lane & 16) ? 8 : 0);
    float badd = bias ? bias[col] : 0.0f;
#pragma unroll
    for (int v = 0; v < 8; ++v) {
      float val = acc[v] + badd;
      if (ACT == 1) val = sp(val);
      if (ACT == 2) val = tanhf(val);
      if (OBF) ((bf16_t*)outp)[(size_t)(row0 + v) * so + col] = f2bf(val);
      else     ((float*)outp)[(size_t)(row0 + v) * so + col]  = val;
    }
  }
}

// ---------------------------------------------------------------------------
// Weight repack: row-major f32 W (N x K) -> bf16 WMMA B-fragment tiles.
// Tile (nt,kt) at (nt*KT+kt)*512; within a tile lane*16+j, where
// n = nt*16 + (lane&15), k = kt*32 + (lane&16?8:0) + (j<8?j:j+8).  K padded w/ 0.
// ---------------------------------------------------------------------------
__global__ void k_prep(const float* __restrict__ W, bf16_t* __restrict__ out,
                       int N, int K, int KT) {
  int e = blockIdx.x * 256 + threadIdx.x;
  int total = (N >> 4) * KT * 512;
  if (e >= total) return;
  int tile = e >> 9, r = e & 511;
  int lane = r >> 4, j = r & 15;
  int kt = tile % KT, nt = tile / KT;
  int n = nt * 16 + (lane & 15);
  int k = kt * 32 + ((lane & 16) ? 8 : 0) + (j < 8 ? j : j + 8);
  float v = (k < K) ? W[(size_t)n * K + k] : 0.f;
  out[e] = f2bf(v);
}

// Same, but B is the K-concat [W0 | W1] (for the GRU [x,h] trick).
__global__ void k_prep_cat(const float* __restrict__ W0, const float* __restrict__ W1,
                           bf16_t* __restrict__ out, int N, int K0, int K1, int KT) {
  int e = blockIdx.x * 256 + threadIdx.x;
  int total = (N >> 4) * KT * 512;
  if (e >= total) return;
  int tile = e >> 9, r = e & 511;
  int lane = r >> 4, j = r & 15;
  int kt = tile % KT, nt = tile / KT;
  int n = nt * 16 + (lane & 15);
  int k = kt * 32 + ((lane & 16) ? 8 : 0) + (j < 8 ? j : j + 8);
  float v = 0.f;
  if (k < K0)            v = W0[(size_t)n * K0 + k];
  else if (k < K0 + K1)  v = W1[(size_t)n * K1 + (k - K0)];
  out[e] = f2bf(v);
}

// ---------------------------------------------------------------------------
// Kernel 1: reverse-time GRU + encoder head. 16 batch rows / block.
// grz = [x,h] @ [Wih_rz;Whh_rz]^T  (one K=256 GEMM), n-gate parts separate.
// ---------------------------------------------------------------------------
__global__ __launch_bounds__(256)
void k_gru(const float* __restrict__ xs,
           const float* __restrict__ bih, const float* __restrict__ bhh,
           const float* __restrict__ enc_b,
           const bf16_t* __restrict__ WrzP, const bf16_t* __restrict__ WinP,
           const bf16_t* __restrict__ WhnP, const bf16_t* __restrict__ encP,
           float* __restrict__ ctx)
{
  __shared__ alignas(16) float  xstage[16 * 128];
  __shared__ alignas(16) bf16_t xh[16 * 256];      // cols 0..127 = x, 128..255 = h
  __shared__ alignas(16) float  hst[16 * 128];
  __shared__ alignas(16) float  grz[16 * 256];
  __shared__ alignas(16) float  ginn[16 * 128];
  __shared__ alignas(16) float  ghn[16 * 128];
  __shared__ alignas(16) float  ctxs[16 * 64];
  int tid = threadIdx.x, wid = tid >> 5, lane = tid & 31;
  int b0 = blockIdx.x * 16;

  for (int i = tid; i < 16 * 128; i += 256) {
    hst[i] = 0.f;
    xh[(i >> 7) * 256 + 128 + (i & 127)] = f2bf(0.f);
  }
  __syncthreads();

  for (int t = T_ - 1; t >= 0; --t) {
    const float* gx = xs + ((size_t)t * B_ + b0) * D_;
    // async-stage x tile (8KB = 512 x 16B) into LDS
    for (int i = tid; i < 512; i += 256)
      async_ld16((unsigned)(uintptr_t)xstage + i * 16, (const char*)gx + i * 16);
    wait_async0();
    __syncthreads();
    if (t > 0)
      __builtin_prefetch(xs + ((size_t)(t - 1) * B_ + b0) * D_ + tid * 8, 0, 1);
    for (int i = tid; i < 16 * 128; i += 256)
      xh[(i >> 7) * 256 + (i & 127)] = f2bf(xstage[i]);
    __syncthreads();

    gemm_lds<1,16,8,0,0>(xh,       256, WrzP, nullptr, grz,  256, wid, lane);
    gemm_lds<1, 8,4,0,0>(xh,       256, WinP, nullptr, ginn, 128, wid, lane);
    gemm_lds<1, 8,4,0,0>(xh + 128, 256, WhnP, nullptr, ghn,  128, wid, lane);
    __syncthreads();

    for (int i = tid; i < 16 * 128; i += 256) {
      int r = i >> 7, j = i & 127;
      float rg = sigm(grz[r * 256 + j]       + bih[j]       + bhh[j]);
      float zg = sigm(grz[r * 256 + 128 + j] + bih[128 + j] + bhh[128 + j]);
      float ng = tanhf(ginn[i] + bih[256 + j] + rg * (ghn[i] + bhh[256 + j]));
      float h  = (1.f - zg) * ng + zg * hst[i];
      hst[i] = h;
      xh[r * 256 + 128 + j] = f2bf(h);
    }
    __syncthreads();

    gemm_lds<1,4,4,0,0>(xh + 128, 256, encP, enc_b, ctxs, 64, wid, lane);
    __syncthreads();
    for (int i = tid; i < 16 * 64; i += 256)
      ctx[((size_t)t * B_ + b0 + (i >> 6)) * C_ + (i & 63)] = ctxs[i];
  }
}

// ---------------------------------------------------------------------------
// Kernel 2: qz0/z0/KL + SDE integration + projector + log-lik partials.
// 32 batch rows / block.
// ---------------------------------------------------------------------------
__global__ __launch_bounds__(256)
void k_sde(const float* __restrict__ xs, const float* __restrict__ noise_std,
           const float* __restrict__ actions, const float* __restrict__ eps0,
           const float* __restrict__ dW,
           const float* __restrict__ qz0_b,
           const float* __restrict__ f_b1, const float* __restrict__ f_b2, const float* __restrict__ f_b3,
           const float* __restrict__ h_b1, const float* __restrict__ h_b2, const float* __restrict__ h_b3,
           const float* __restrict__ gW1, const float* __restrict__ gb1,
           const float* __restrict__ gW2, const float* __restrict__ gb2,
           const float* __restrict__ p_b1, const float* __restrict__ p_b2, const float* __restrict__ p_b3,
           const float* __restrict__ ae_b,
           const float* __restrict__ pz0m, const float* __restrict__ pz0ls,
           const bf16_t* __restrict__ qz0P,
           const bf16_t* __restrict__ f1P, const bf16_t* __restrict__ f2P, const bf16_t* __restrict__ f3P,
           const bf16_t* __restrict__ h1P, const bf16_t* __restrict__ h2P, const bf16_t* __restrict__ h3P,
           const bf16_t* __restrict__ p1P, const bf16_t* __restrict__ p2P, const bf16_t* __restrict__ p3P,
           const bf16_t* __restrict__ aeP,
           const float* __restrict__ ctx,
           float* __restrict__ lpP, float* __restrict__ klP, float* __restrict__ lrP)
{
  __shared__ alignas(16) float  zb[32 * 16];
  __shared__ alignas(16) bf16_t zpad[32 * 32];     // z | zeros (K pad for 16->128 GEMMs)
  __shared__ alignas(16) bf16_t catA[32 * 96];     // [z(16) | c(64) | pad(16)]
  __shared__ alignas(16) bf16_t mh1[32 * 128];
  __shared__ alignas(16) bf16_t mh2[32 * 128];
  __shared__ alignas(16) float  f1o[512], f2o[512], hdo[512], g1o[512], g2o[512], ztb[512];
  __shared__ alignas(16) bf16_t xprevb[32 * 128];
  __shared__ alignas(16) char   pool[16384];       // aeA (bf16 32x160) / xout|qzo (f32 32x128)
  bf16_t* aeA  = (bf16_t*)pool;
  float*  xout = (float*)pool;
  float*  qzo  = (float*)pool;

  int tid = threadIdx.x, wid = tid >> 5, lane = tid & 31;
  int b0 = blockIdx.x * 32;

  float accLP = 0.f, accKL = 0.f, accLR = 0.f;
  float ns = noise_std[0];
  float inv_ns = 1.f / ns;
  float lpc = -logf(ns) - 0.5f * LOG2PI_;

  for (int i = tid; i < 32 * 32; i += 256) zpad[i] = f2bf(0.f);
  for (int i = tid; i < 32 * 96; i += 256) catA[i] = f2bf(0.f);
  __syncthreads();

  // ---- qz0 from ctx[0] -------------------------------------------------
  for (int i = tid; i < 32 * 64; i += 256)
    mh1[i] = f2bf(ctx[((size_t)(b0 + (i >> 6))) * C_ + (i & 63)]);
  __syncthreads();
  gemm_lds<2,2,2,0,0>(mh1, 64, qz0P, qz0_b, qzo, 32, wid, lane);
  __syncthreads();
  for (int i = tid; i < 512; i += 256) {
    int r = i >> 4, l = i & 15;
    float qm = qzo[r * 32 + l], qls = qzo[r * 32 + 16 + l];
    float z0 = qm + expf(qls) * eps0[(size_t)(b0 + r) * L_ + l];
    float pm = pz0m[l], pls = pz0ls[l];
    float dm = qm - pm;
    accKL += pls - qls + (expf(2.f * qls) + dm * dm) / (2.f * expf(2.f * pls)) - 0.5f;
    zb[i] = z0;
    zpad[r * 32 + l] = f2bf(z0);
  }
  for (int i = tid; i < 32 * 128; i += 256)
    xprevb[i] = f2bf(xs[((size_t)(b0 + (i >> 7))) * D_ + (i & 127)]);
  __syncthreads();

  // ---- projector eval (zpad -> xout) + optional log-lik & x_prev -------
  auto proj = [&](int t_idx, bool toXprev) {
    gemm_lds<2,8,1,2,1>(zpad, 32,  p1P, p_b1, mh1, 128, wid, lane); __syncthreads();
    gemm_lds<2,8,4,0,1>(mh1, 128,  p2P, p_b2, mh2, 128, wid, lane); __syncthreads();
    gemm_lds<2,8,4,0,0>(mh2, 128,  p3P, p_b3, xout, 128, wid, lane); __syncthreads();
    if (t_idx >= 0) {
      const float* xr = xs + ((size_t)t_idx * B_ + b0) * D_;
      for (int i = tid; i < 32 * 128; i += 256) {
        float diff = (xr[i] - xout[i]) * inv_ns;
        accLP += lpc - 0.5f * diff * diff;
      }
    }
    if (toXprev)
      for (int i = tid; i < 32 * 128; i += 256) xprevb[i] = f2bf(xout[i]);
    __syncthreads();
  };
  auto gdiff = [&](const float* zsrc, float* gout) {
    for (int i = tid; i < 512; i += 256) {
      int l = i & 15;
      float y = zsrc[i], gs = 0.f;
      for (int hh = 0; hh < 128; ++hh)
        gs += sp(fmaf(y, gW1[l * 128 + hh], gb1[l * 128 + hh])) * gW2[l * 128 + hh];
      gout[i] = sigm(gs + gb2[l]);
    }
  };

  proj(0, false);  // x0 vs xs[0]

  // ---- segments ---------------------------------------------------------
  for (int seg = 0; seg < NSEG_; ++seg) {
    int t0 = seg * SKIP_;
    for (int i = tid; i < 32 * 64; i += 256) {
      int r = i >> 6, c = i & 63;
      catA[r * 96 + 16 + c] = f2bf(ctx[((size_t)t0 * B_ + b0 + r) * C_ + c]);
    }
    for (int i = tid; i < 32 * 160; i += 256) {
      int r = i / 160, c = i % 160;
      bf16_t v;
      if (c < 16)       v = f2bf(zb[r * 16 + c]);
      else if (c < 24)  v = f2bf(actions[((size_t)t0 * B_ + b0 + r) * A_ + (c - 16)]);
      else if (c < 152) v = xprevb[r * 128 + (c - 24)];
      else              v = f2bf(0.f);
      aeA[i] = v;
    }
    __syncthreads();
    gemm_lds<2,1,5,0,0>(aeA, 160, aeP, ae_b, zb, 16, wid, lane);
    __syncthreads();
    for (int i = tid; i < 512; i += 256) {
      int r = i >> 4, l = i & 15;
      bf16_t zv = f2bf(zb[i]);
      zpad[r * 32 + l] = zv;
      catA[r * 96 + l] = zv;
    }
    __syncthreads();
    proj(t0 + 1, false);  // x_seg[0]

    for (int k = 0; k < NSTEPS_; ++k) {
      const float* dwp = dW + (((size_t)seg * NSTEPS_ + k) * B_ + b0) * L_;
      // f1 = f_drift(z, c)
      gemm_lds<2,8,3,1,1>(catA, 96, f1P, f_b1, mh1, 128, wid, lane); __syncthreads();
      gemm_lds<2,8,4,1,1>(mh1, 128, f2P, f_b2, mh2, 128, wid, lane); __syncthreads();
      gemm_lds<2,1,4,0,0>(mh2, 128, f3P, f_b3, f1o, 16, wid, lane);
      // hd = h_drift(z)
      gemm_lds<2,8,1,1,1>(zpad, 32, h1P, h_b1, mh1, 128, wid, lane); __syncthreads();
      gemm_lds<2,8,4,1,1>(mh1, 128, h2P, h_b2, mh2, 128, wid, lane); __syncthreads();
      gemm_lds<2,1,4,0,0>(mh2, 128, h3P, h_b3, hdo, 16, wid, lane);
      gdiff(zb, g1o);
      __syncthreads();
      // zt = z + f1*dt + g1*dw
      for (int i = tid; i < 512; i += 256) {
        int r = i >> 4, l = i & 15;
        float dwk = dwp[r * L_ + l] * SQRTDT_;
        float zt = zb[i] + f1o[i] * DT_ + g1o[i] * dwk;
        ztb[i] = zt;
        catA[r * 96 + l] = f2bf(zt);
      }
      __syncthreads();
      // f2 = f_drift(zt, c)
      gemm_lds<2,8,3,1,1>(catA, 96, f1P, f_b1, mh1, 128, wid, lane); __syncthreads();
      gemm_lds<2,8,4,1,1>(mh1, 128, f2P, f_b2, mh2, 128, wid, lane); __syncthreads();
      gemm_lds<2,1,4,0,0>(mh2, 128, f3P, f_b3, f2o, 16, wid, lane);
      gdiff(ztb, g2o);
      __syncthreads();
      // Heun update + path KL
      for (int i = tid; i < 512; i += 256) {
        int r = i >> 4, l = i & 15;
        float dwk = dwp[r * L_ + l] * SQRTDT_;
        float zn = zb[i] + 0.5f * (f1o[i] + f2o[i]) * DT_ + 0.5f * (g1o[i] + g2o[i]) * dwk;
        float u = (f1o[i] - hdo[i]) / g1o[i];
        accLR += 0.5f * u * u * DT_;
        zb[i] = zn;
        bf16_t zv = f2bf(zn);
        zpad[r * 32 + l] = zv;
        catA[r * 96 + l] = zv;
      }
      __syncthreads();
      if (k & 1) {
        int tp = t0 + 2 + (k >> 1);
        proj(tp < T_ ? tp : -1, k == NSTEPS_ - 1);
      }
    }
  }

  // ---- deterministic block reductions -----------------------------------
  __syncthreads();
  auto reduce = [&](float v) -> float {
    ztb[tid] = v; __syncthreads();
    for (int s = 128; s > 0; s >>= 1) {
      if (tid < s) ztb[tid] += ztb[tid + s];
      __syncthreads();
    }
    float r = ztb[0]; __syncthreads(); return r;
  };
  float lpS = reduce(accLP);
  float klS = reduce(accKL);
  float lrS = reduce(accLR);
  if (tid == 0) { lpP[blockIdx.x] = lpS; klP[blockIdx.x] = klS; lrP[blockIdx.x] = lrS; }
}

__global__ void k_final(const float* __restrict__ lpP, const float* __restrict__ klP,
                        const float* __restrict__ lrP, int nb, float* __restrict__ out) {
  if (threadIdx.x == 0 && blockIdx.x == 0) {
    float a = 0.f, b = 0.f, c = 0.f;
    for (int i = 0; i < nb; ++i) { a += lpP[i]; b += klP[i]; c += lrP[i]; }
    out[0] = a / (float)B_;
    out[1] = b / (float)B_ + c / (float)B_;
  }
}

// ---------------------------------------------------------------------------
// Host launcher
// ---------------------------------------------------------------------------
extern "C" void kernel_launch(void* const* d_in, const int* in_sizes, int n_in,
                              void* d_out, int out_size, void* d_ws, size_t ws_size,
                              hipStream_t stream) {
  (void)in_sizes; (void)n_in; (void)out_size; (void)ws_size;
  const float* xs        = (const float*)d_in[0];
  const float* noise_std = (const float*)d_in[2];
  const float* actions   = (const float*)d_in[3];
  const float* eps0      = (const float*)d_in[4];
  const float* dW        = (const float*)d_in[5];
  const float* gru_Wih   = (const float*)d_in[6];
  const float* gru_Whh   = (const float*)d_in[7];
  const float* gru_bih   = (const float*)d_in[8];
  const float* gru_bhh   = (const float*)d_in[9];
  const float* enc_W     = (const float*)d_in[10];
  const float* enc_b     = (const float*)d_in[11];
  const float* qz0_W     = (const float*)d_in[12];
  const float* qz0_b     = (const float*)d_in[13];
  const float* f_W1      = (const float*)d_in[14];
  const float* f_b1      = (const float*)d_in[15];
  const float* f_W2      = (const float*)d_in[16];
  const float* f_b2      = (const float*)d_in[17];
  const float* f_W3      = (const float*)d_in[18];
  const float* f_b3      = (const float*)d_in[19];
  const float* h_W1      = (const float*)d_in[20];
  const float* h_b1      = (const float*)d_in[21];
  const float* h_W2      = (const float*)d_in[22];
  const float* h_b2      = (const float*)d_in[23];
  const float* h_W3      = (const float*)d_in[24];
  const float* h_b3      = (const float*)d_in[25];
  const float* g_W1      = (const float*)d_in[26];
  const float* g_b1      = (const float*)d_in[27];
  const float* g_W2      = (const float*)d_in[28];
  const float* g_b2      = (const float*)d_in[29];
  const float* proj_W1   = (const float*)d_in[30];
  const float* proj_b1   = (const float*)d_in[31];
  const float* proj_W2   = (const float*)d_in[32];
  const float* proj_b2   = (const float*)d_in[33];
  const float* proj_W3   = (const float*)d_in[34];
  const float* proj_b3   = (const float*)d_in[35];
  const float* ae_W      = (const float*)d_in[36];
  const float* ae_b      = (const float*)d_in[37];
  const float* pz0_mean  = (const float*)d_in[38];
  const float* pz0_lstd  = (const float*)d_in[39];

  char* ws = (char*)d_ws;
  size_t off = 0;
  float* ctx = (float*)ws;  off += (size_t)T_ * B_ * C_ * sizeof(float);
  auto abf = [&](size_t elems) { bf16_t* p = (bf16_t*)(ws + off); off += elems * sizeof(bf16_t); return p; };
  bf16_t* WrzP = abf(65536);
  bf16_t* WinP = abf(16384);
  bf16_t* WhnP = abf(16384);
  bf16_t* encP = abf(8192);
  bf16_t* qz0P = abf(2048);
  bf16_t* f1P  = abf(12288);
  bf16_t* f2P  = abf(16384);
  bf16_t* f3P  = abf(2048);
  bf16_t* h1P  = abf(4096);
  bf16_t* h2P  = abf(16384);
  bf16_t* h3P  = abf(2048);
  bf16_t* p1P  = abf(4096);
  bf16_t* p2P  = abf(16384);
  bf16_t* p3P  = abf(16384);
  bf16_t* aeP  = abf(2560);
  float* lpP = (float*)(ws + off); off += 64 * sizeof(float);
  float* klP = (float*)(ws + off); off += 64 * sizeof(float);
  float* lrP = (float*)(ws + off); off += 64 * sizeof(float);

  auto prep = [&](const float* W, bf16_t* o, int N, int K, int KT) {
    int total = (N / 16) * KT * 512;
    k_prep<<<(total + 255) / 256, 256, 0, stream>>>(W, o, N, K, KT);
  };
  {
    int total = (256 / 16) * 8 * 512;
    k_prep_cat<<<(total + 255) / 256, 256, 0, stream>>>(gru_Wih, gru_Whh, WrzP, 256, 128, 128, 8);
  }
  prep(gru_Wih + 256 * 128, WinP, 128, 128, 4);
  prep(gru_Whh + 256 * 128, WhnP, 128, 128, 4);
  prep(enc_W,   encP, 64, 128, 4);
  prep(qz0_W,   qz0P, 32, 64, 2);
  prep(f_W1,    f1P, 128, 80, 3);
  prep(f_W2,    f2P, 128, 128, 4);
  prep(f_W3,    f3P, 16, 128, 4);
  prep(h_W1,    h1P, 128, 16, 1);
  prep(h_W2,    h2P, 128, 128, 4);
  prep(h_W3,    h3P, 16, 128, 4);
  prep(proj_W1, p1P, 128, 16, 1);
  prep(proj_W2, p2P, 128, 128, 4);
  prep(proj_W3, p3P, 128, 128, 4);
  prep(ae_W,    aeP, 16, 152, 5);

  k_gru<<<B_ / 16, 256, 0, stream>>>(xs, gru_bih, gru_bhh, enc_b,
                                     WrzP, WinP, WhnP, encP, ctx);

  k_sde<<<B_ / 32, 256, 0, stream>>>(xs, noise_std, actions, eps0, dW,
                                     qz0_b, f_b1, f_b2, f_b3,
                                     h_b1, h_b2, h_b3,
                                     g_W1, g_b1, g_W2, g_b2,
                                     proj_b1, proj_b2, proj_b3,
                                     ae_b, pz0_mean, pz0_lstd,
                                     qz0P, f1P, f2P, f3P,
                                     h1P, h2P, h3P,
                                     p1P, p2P, p3P, aeP,
                                     ctx, lpP, klP, lrP);

  k_final<<<1, 32, 0, stream>>>(lpP, klP, lrP, B_ / 32, (float*)d_out);
}
